// Dominant_87282325390067
// MI455X (gfx1250) — matrix operand order repaired and checked by reading, not compiled, not run on hardware
//
#include <hip/hip_runtime.h>

// ---------------------------------------------------------------------------
// DOMINANT graph autoencoder (6x SAGEConv) for MI455X / gfx1250.
// GEMMs run as bf16 WMMA (v_wmma_f32_16x16x32_bf16, f32 accumulate).
// The final [N,N].T output is produced transpose-free by swapping GEMM
// operands (s = W'^T-as-rows  x  catA^T), so the 400MB store is coalesced.
// ---------------------------------------------------------------------------

typedef __attribute__((ext_vector_type(16))) __bf16 v16bf;
typedef __attribute__((ext_vector_type(8)))  __bf16 v8bf;
typedef __attribute__((ext_vector_type(8)))  float  v8f;

static constexpr int   kN = 10000;   // nodes
static constexpr int   kE = 320000;  // edges
static constexpr int   kF = 128;     // feature width (all layer inputs are 128)
static constexpr int   kK = 256;     // fused GEMM K = 2*128  ([agg ; x])

// ------------------------------ degree ------------------------------------
__global__ void deg_kernel(const int* __restrict__ dst, float* __restrict__ deg, int E) {
  int e = blockIdx.x * blockDim.x + threadIdx.x;
  if (e < E) atomicAdd(&deg[dst[e]], 1.0f);
}

__global__ void rdeg_kernel(float* __restrict__ deg, int n) {
  int i = blockIdx.x * blockDim.x + threadIdx.x;
  if (i < n) deg[i] = 1.0f / fmaxf(deg[i], 1.0f);
}

// --------------------------- mean aggregation ------------------------------
// one wave32 per edge; lane handles one float4 chunk of the 128-wide feature
__global__ void aggregate_kernel(const float* __restrict__ x,
                                 const int* __restrict__ src,
                                 const int* __restrict__ dst,
                                 float* __restrict__ msg, int E) {
  int warp = (blockIdx.x * blockDim.x + threadIdx.x) >> 5;
  int lane = threadIdx.x & 31;
  if (warp >= E) return;
  int s = src[warp];
  int d = dst[warp];
  const float4 v = *(const float4*)(x + (size_t)s * kF + lane * 4);
  float* o = msg + (size_t)d * kF + lane * 4;
  atomicAdd(o + 0, v.x);
  atomicAdd(o + 1, v.y);
  atomicAdd(o + 2, v.z);
  atomicAdd(o + 3, v.w);
}

// ------------------- build bf16 activation [mean_agg ; x] ------------------
__global__ void build_catA(const float* __restrict__ act,
                           const float* __restrict__ msg,
                           const float* __restrict__ rdeg,
                           __bf16* __restrict__ catA, int n_nodes) {
  int t = blockIdx.x * blockDim.x + threadIdx.x;      // n_nodes * 64 threads
  if (t >= n_nodes * 64) return;
  int n = t >> 6;
  int q = (t & 63) * 4;                               // k offset 0..252
  float4 v;
  if (q < kF) {
    float r = rdeg[n];
    v = *(const float4*)(msg + (size_t)n * kF + q);
    v.x *= r; v.y *= r; v.z *= r; v.w *= r;
  } else {
    v = *(const float4*)(act + (size_t)n * kF + (q - kF));
  }
  __bf16* o = catA + (size_t)n * kK + q;
  o[0] = (__bf16)v.x; o[1] = (__bf16)v.y; o[2] = (__bf16)v.z; o[3] = (__bf16)v.w;
}

// ---------- build bf16 transposed fused weight  catWT[n][k]=[Wl;Wr][k][n] --
__global__ void build_catWT(const float* __restrict__ Wl,
                            const float* __restrict__ Wr,
                            __bf16* __restrict__ catWT, int dout) {
  int t = blockIdx.x * blockDim.x + threadIdx.x;      // dout * 256 threads
  if (t >= dout * kK) return;
  int n = t % dout;                                   // coalesced W reads
  int k = t / dout;
  float v = (k < kF) ? Wl[(size_t)k * dout + n] : Wr[(size_t)(k - kF) * dout + n];
  catWT[(size_t)n * kK + k] = (__bf16)v;
}

// ------------------------------- WMMA GEMM ---------------------------------
// C[m][n] = relu( sum_k A[m][k]*B[n][k] + bias )      (A: MxK, B: NoutxK, NT)
// block = 256 threads (8 wave32), tile 64x128, BK=32, K fixed at 256.
#define BM 64
#define BN 128
#define BK 32
#define APAD 40   // bf16 elems per LDS row (80B) -> staggers 64 banks

__global__ __launch_bounds__(256) void gemm_bf16_relu(
    const __bf16* __restrict__ A, const __bf16* __restrict__ B,
    const float* __restrict__ bias, float* __restrict__ C,
    int M, int Nout, int ldC, int bias_row) {
  __shared__ __bf16 sA[BM * APAD];
  __shared__ __bf16 sB[BN * APAD];

  const int tid  = threadIdx.x;
  const int m0   = blockIdx.x * BM;
  const int n0   = blockIdx.y * BN;
  const int wave = tid >> 5;
  const int lane = tid & 31;
  const int rw   = wave & 3;        // 16-row sub-tile within 64
  const int cw   = wave >> 2;       // 64-col half within 128
  const int half = lane >> 4;
  const int l15  = lane & 15;

  v8f acc[4];
#pragma unroll
  for (int j = 0; j < 4; ++j)
#pragma unroll
    for (int r = 0; r < 8; ++r) acc[j][r] = 0.0f;

  const int lr = tid >> 2;          // 0..63  (row / col for LDS fill)
  const int lc = tid & 3;           // 8-bf16 chunk within BK

  for (int ks = 0; ks < kK; ks += BK) {
    __syncthreads();
    { // A tile: 64x32
      int gr = m0 + lr;
      v8bf v{};
      if (gr < M) {
        const __bf16* p = A + (size_t)gr * kK + ks + lc * 8;
        v = *(const v8bf*)p;
        if (ks + BK < kK) __builtin_prefetch(p + BK, 0, 3);  // global_prefetch_b8
      }
      *(v8bf*)(sA + lr * APAD + lc * 8) = v;
    }
#pragma unroll
    for (int h = 0; h < 2; ++h) { // B tile: 128x32, K-contiguous per column
      int c  = lr + h * 64;
      int gc = n0 + c;
      v8bf v{};
      if (gc < Nout) v = *(const v8bf*)(B + (size_t)gc * kK + ks + lc * 8);
      *(v8bf*)(sB + c * APAD + lc * 8) = v;
    }
    __syncthreads();

    // A fragment: row = 16*rw + l15 ; K = {half*8..+7} U {16+half*8..+7}
    const __bf16* ap = sA + (16 * rw + l15) * APAD + half * 8;
    v8bf a_lo = *(const v8bf*)(ap);
    v8bf a_hi = *(const v8bf*)(ap + 16);
    v16bf a = __builtin_shufflevector(a_lo, a_hi,
        0, 1, 2, 3, 4, 5, 6, 7, 8, 9, 10, 11, 12, 13, 14, 15);

#pragma unroll
    for (int j = 0; j < 4; ++j) {
      // B fragment: col = 64*cw + 16*j + l15 ; K = {half*8..+7} U {16+half*8..+7}
      const __bf16* bp = sB + (64 * cw + 16 * j + l15) * APAD + half * 8;
      v8bf b_lo = *(const v8bf*)(bp);
      v8bf b_hi = *(const v8bf*)(bp + 16);
      v16bf b = __builtin_shufflevector(b_lo, b_hi,
          0, 1, 2, 3, 4, 5, 6, 7, 8, 9, 10, 11, 12, 13, 14, 15);
      acc[j] = __builtin_amdgcn_wmma_f32_16x16x32_bf16(
          /*neg_a=*/false, a, /*neg_b=*/false, b,
          /*c_mod=*/(short)0, acc[j], /*reuse_a=*/false, /*reuse_b=*/false);
    }
  }

  // epilogue: D layout — VGPR r -> M = r + 8*half, N = l15
#pragma unroll
  for (int j = 0; j < 4; ++j) {
    int col = n0 + 64 * cw + 16 * j + l15;
    if (col >= Nout) continue;
#pragma unroll
    for (int r = 0; r < 8; ++r) {
      int row = m0 + 16 * rw + half * 8 + r;
      if (row >= M) continue;
      float v = acc[j][r] + (bias_row ? bias[row] : bias[col]);
      C[(size_t)row * ldC + col] = v > 0.0f ? v : 0.0f;
    }
  }
}

// ------------------------------ orchestration ------------------------------
extern "C" void kernel_launch(void* const* d_in, const int* in_sizes, int n_in,
                              void* d_out, int out_size, void* d_ws, size_t ws_size,
                              hipStream_t stream) {
  const float* x   = (const float*)d_in[0];
  const int*   ei  = (const int*)d_in[1];
  const int*   src = ei;
  const int*   dst = ei + kE;

  const float *Wl[6], *bl[6], *Wr[6];
  for (int i = 0; i < 6; ++i) {
    Wl[i] = (const float*)d_in[2 + 3 * i];
    bl[i] = (const float*)d_in[3 + 3 * i];
    Wr[i] = (const float*)d_in[4 + 3 * i];
  }

  float* out_s  = (float*)d_out;                       // [N, N]
  float* out_xh = out_s + (size_t)kN * kN;             // [N, F]

  // workspace carve (~26 MB total)
  char* w = (char*)d_ws;
  auto carve = [&](size_t bytes) {
    char* p = w;
    w += (bytes + 255) & ~(size_t)255;
    return p;
  };
  float*  rdeg  = (float*)carve((size_t)kN * 4);
  float*  msg   = (float*)carve((size_t)kN * kF * 4);
  __bf16* catA  = (__bf16*)carve((size_t)kN * kK * 2);
  __bf16* catWT = (__bf16*)carve((size_t)kN * kK * 2);  // max dout = N
  float*  act1  = (float*)carve((size_t)kN * kF * 4);
  float*  act2  = (float*)carve((size_t)kN * kF * 4);

  // degree reciprocal (graph-constant, reused by all 6 layers)
  hipMemsetAsync(rdeg, 0, (size_t)kN * 4, stream);
  deg_kernel<<<(kE + 255) / 256, 256, 0, stream>>>(dst, rdeg, kE);
  rdeg_kernel<<<(kN + 255) / 256, 256, 0, stream>>>(rdeg, kN);

  auto layer = [&](const float* in, int li, int dout, float* outp, int ldC,
                   bool transposed_out) {
    hipMemsetAsync(msg, 0, (size_t)kN * kF * 4, stream);
    aggregate_kernel<<<(kE * 32 + 255) / 256, 256, 0, stream>>>(in, src, dst, msg, kE);
    build_catA<<<(kN * 64 + 255) / 256, 256, 0, stream>>>(in, msg, rdeg, catA, kN);
    build_catWT<<<(dout * kK + 255) / 256, 256, 0, stream>>>(Wl[li], Wr[li], catWT, dout);
    if (!transposed_out) {
      dim3 grid((kN + BM - 1) / BM, (dout + BN - 1) / BN);
      gemm_bf16_relu<<<grid, 256, 0, stream>>>(catA, catWT, bl[li], outp,
                                               kN, dout, ldC, 0);
    } else {
      // s = (catA @ catW)^T computed directly: rows = weight columns.
      dim3 grid((dout + BM - 1) / BM, (kN + BN - 1) / BN);
      gemm_bf16_relu<<<grid, 256, 0, stream>>>(catWT, catA, bl[li], outp,
                                               dout, kN, ldC, 1);
    }
  };

  layer(x,    0, kF, act1,   kF, false);  // e1
  layer(act1, 1, kF, act2,   kF, false);  // e2   (h kept in act2)
  layer(act2, 2, kF, act1,   kF, false);  // a1
  layer(act1, 3, kF, out_xh, kF, false);  // a2 -> xh
  layer(act2, 4, kF, act1,   kF, false);  // s1
  layer(act1, 5, kN, out_s,  kN, true);   // s2 -> s (transpose-free)
}
